// GNN_15753940042144
// MI455X (gfx1250) — compile-verified, matrix operand order checked by reading, not run on hardware
//
#include <hip/hip_runtime.h>
#include <hip/hip_bf16.h>

typedef __bf16 bf16;
typedef __attribute__((ext_vector_type(16))) __bf16 v16bf;
typedef __attribute__((ext_vector_type(8)))  __bf16 v8bf;
typedef __attribute__((ext_vector_type(8)))  float  v8f;
typedef __attribute__((ext_vector_type(4)))  float  v4f;
typedef __attribute__((ext_vector_type(4)))  unsigned v4u;
typedef __attribute__((ext_vector_type(8)))  int    v8i;
typedef __attribute__((ext_vector_type(4)))  int    v4i;

#define N_NODES 50000
#define N_EDGES 800000
#define IN_C 512
#define HID_C 256
#define OUT_C 128

// ---------------- degree / normalization ----------------
__global__ void k_init_deg(float* __restrict__ deg, int n) {
    int i = blockIdx.x * blockDim.x + threadIdx.x;
    if (i < n) deg[i] = 1.0f;            // self-loop contributes 1
}

__global__ void k_accum_deg(const long long* __restrict__ dst, float* __restrict__ deg, int ne) {
    int e = blockIdx.x * blockDim.x + threadIdx.x;
    if (e < ne) atomicAdd(&deg[(int)dst[e]], 1.0f);
}

__global__ void k_finish_deg(float* __restrict__ deg, int n) {
    int i = blockIdx.x * blockDim.x + threadIdx.x;
    if (i < n) deg[i] = __frsqrt_rn(fmaxf(deg[i], 1.0f));   // dinv
}

// ---------------- conversions ----------------
__global__ void k_cvt_f32_bf16(const float* __restrict__ in, bf16* __restrict__ out, long long n4) {
    long long i = (long long)blockIdx.x * blockDim.x + threadIdx.x;
    if (i >= n4) return;
    v4f v = *(const v4f*)(in + i * 4);
    bf16* o = out + i * 4;
    o[0] = (bf16)v.x; o[1] = (bf16)v.y; o[2] = (bf16)v.z; o[3] = (bf16)v.w;
}

// W is K x N row-major (f32); Wt is N x K row-major (bf16) == W^T
__global__ void k_transpose_bf16(const float* __restrict__ W, bf16* __restrict__ Wt,
                                 int K, int N) {
    int g = blockIdx.x * blockDim.x + threadIdx.x;
    if (g >= K * N) return;
    int k = g / N, n = g - k * N;
    Wt[(size_t)n * K + k] = (bf16)W[g];
}

__global__ void k_zero_f32(float* __restrict__ p, long long n4) {
    long long i = (long long)blockIdx.x * blockDim.x + threadIdx.x;
    if (i >= n4) return;
    v4f z = {0.f, 0.f, 0.f, 0.f};
    *(v4f*)(p + i * 4) = z;
}

// ---------------- WMMA GEMM: C[M,N] = A[M,K](bf16) * Bt[N,K](bf16)^T ----------------
// One wave -> 16 rows x 64 cols (4 WMMA tiles). Block = 8 waves = 128 rows.
// The block's 64xK bf16 weight panel is DMA'd into LDS once by the TDM.
__global__ __launch_bounds__(256) void k_gemm_bf16_wmma(
    const bf16* __restrict__ A, const bf16* __restrict__ Bt,
    float* __restrict__ Cf, bf16* __restrict__ Cb,
    int M, int N, int K)
{
    __shared__ bf16 bsm[64 * 512];        // 64 cols x K (K<=512) = 64KB max

    const int wave = threadIdx.x >> 5;
    const int lane = threadIdx.x & 31;
    const int m0 = (blockIdx.x * 8 + wave) * 16;
    const int n0 = blockIdx.y * 64;

    // ---- TDM: stage weight panel Bt[n0 .. n0+63][0..K) into LDS ----
    if (wave == 0) {
        unsigned long long ga = (unsigned long long)(const void*)(Bt + (size_t)n0 * K);
        unsigned lds_off = (unsigned)(unsigned long long)(uintptr_t)&bsm[0]; // low 32 bits of the LDS pointer
        v4u g0;
        g0[0] = 1u;                                  // count=1 valid user descriptor
        g0[1] = lds_off;                             // lds_addr (bytes)
        g0[2] = (unsigned)(ga & 0xFFFFFFFFu);        // global_addr[31:0]
        g0[3] = (unsigned)((ga >> 32) & 0x01FFFFFFu) // global_addr[56:32]
                | 0x80000000u;                       // type=2 ("image")
        v8i g1;
        g1[0] = (int)(1u << 16);                     // data_size=1 (2B); no multicast/pad/iterate
        g1[1] = (int)((unsigned)K << 16);            // tensor_dim0[15:0] in [31:16]
        g1[2] = (int)(64u << 16);                    // tensor_dim0 hi=0; tensor_dim1=64
        g1[3] = (int)((unsigned)K << 16);            // tensor_dim1 hi=0; tile_dim0=K
        g1[4] = 64;                                  // tile_dim1=64; tile_dim2=0
        g1[5] = K;                                   // tensor_dim0_stride[31:0] = K
        g1[6] = 0;                                   // stride hi / dim1_stride lo
        g1[7] = 0;                                   // dim1_stride hi
        v4i g2 = {0, 0, 0, 0};
        v4i g3 = {0, 0, 0, 0};
        v8i g4 = {0, 0, 0, 0, 0, 0, 0, 0};
        __builtin_amdgcn_tensor_load_to_lds(g0, g1, g2, g3, g4, 0);
        __builtin_amdgcn_s_wait_tensorcnt(0);
    }
    __syncthreads();

    if (m0 >= M) return;                  // M % 16 == 0 -> wave-uniform guard
    const int r15 = lane & 15;            // A row / B col within tile
    const int kh  = (lane >> 4) << 3;     // 0 or 8: K-half select per ISA layout

    v8f acc[4] = {};
    const bf16* arow = A + (size_t)(m0 + r15) * K;

    for (int kk = 0; kk < K; kk += 32) {
        v16bf a;
        {
            v8bf c0 = *(const v8bf*)(arow + kk + kh);
            v8bf c1 = *(const v8bf*)(arow + kk + kh + 16);
            #pragma unroll
            for (int i = 0; i < 8; ++i) { a[i] = c0[i]; a[8 + i] = c1[i]; }
        }
        #pragma unroll
        for (int t = 0; t < 4; ++t) {
            const bf16* bcol = &bsm[(size_t)(t * 16 + r15) * K + kk + kh];  // LDS reads
            v16bf b;
            v8bf c0 = *(const v8bf*)(bcol);
            v8bf c1 = *(const v8bf*)(bcol + 16);
            #pragma unroll
            for (int i = 0; i < 8; ++i) { b[i] = c0[i]; b[8 + i] = c1[i]; }
            acc[t] = __builtin_amdgcn_wmma_f32_16x16x32_bf16(
                false, a, false, b, (short)0, acc[t], false, false);
        }
    }

    // D layout: VGPR r, lanes 0-15 -> M=r, lanes 16-31 -> M=8+r; N = lane&15
    const int mrow = m0 + ((lane >> 4) << 3);
    const int ncol = lane & 15;
    if (Cb) {
        #pragma unroll
        for (int t = 0; t < 4; ++t)
            #pragma unroll
            for (int r = 0; r < 8; ++r)
                Cb[(size_t)(mrow + r) * N + n0 + t * 16 + ncol] = (bf16)acc[t][r];
    } else {
        #pragma unroll
        for (int t = 0; t < 4; ++t)
            #pragma unroll
            for (int r = 0; r < 8; ++r)
                Cf[(size_t)(mrow + r) * N + n0 + t * 16 + ncol] = acc[t][r];
    }
}

// ---------------- edge scatter: agg[dst] += dinv[s]*dinv[d] * h[src] ----------------
// thread = (edge, 8-feature chunk); cshift = log2(F/8)
__global__ void k_scatter(const long long* __restrict__ src, const long long* __restrict__ dst,
                          const float* __restrict__ dinv, const bf16* __restrict__ h,
                          float* __restrict__ agg, int ne, int F, int cshift)
{
    int g = blockIdx.x * blockDim.x + threadIdx.x;
    if (g >= (ne << cshift)) return;
    int e = g >> cshift;
    int c = (g & ((1 << cshift) - 1)) << 3;
    int s = (int)src[e], d = (int)dst[e];
    float w = dinv[s] * dinv[d];
    v8bf hv = *(const v8bf*)(h + (size_t)s * F + c);
    float* out = agg + (size_t)d * F + c;
    #pragma unroll
    for (int i = 0; i < 8; ++i) atomicAdd(out + i, w * (float)hv[i]);
}

// ---------------- epilogues (self-loop + bias [+ relu]) ----------------
// layer1: h <- bf16(relu(agg + dinv^2*h + b)) in place (h is bf16)
__global__ void k_final_relu(const float* __restrict__ agg, bf16* __restrict__ h,
                             const float* __restrict__ bias, const float* __restrict__ dinv,
                             int F, int fshift, long long total)
{
    long long g = (long long)blockIdx.x * blockDim.x + threadIdx.x;
    if (g >= total) return;
    int node = (int)(g >> fshift);
    int f = (int)(g & (F - 1));
    float di = dinv[node];
    float v = agg[g] + di * di * (float)h[g] + bias[f];
    h[g] = (bf16)fmaxf(v, 0.0f);
}

// layer2: out <- out(=agg) + dinv^2*h2 + b  (f32)
__global__ void k_final_out(float* __restrict__ out, const float* __restrict__ h2,
                            const float* __restrict__ bias, const float* __restrict__ dinv,
                            int F, int fshift, long long total)
{
    long long g = (long long)blockIdx.x * blockDim.x + threadIdx.x;
    if (g >= total) return;
    int node = (int)(g >> fshift);
    int f = (int)(g & (F - 1));
    float di = dinv[node];
    out[g] = out[g] + di * di * h2[g] + bias[f];
}

extern "C" void kernel_launch(void* const* d_in, const int* in_sizes, int n_in,
                              void* d_out, int out_size, void* d_ws, size_t ws_size,
                              hipStream_t stream) {
    const float*     x  = (const float*)d_in[0];
    const long long* ei = (const long long*)d_in[1];   // (2, E) int64
    const float*     W1 = (const float*)d_in[2];
    const float*     b1 = (const float*)d_in[3];
    const float*     W2 = (const float*)d_in[4];
    const float*     b2 = (const float*)d_in[5];
    const long long* src = ei;
    const long long* dst = ei + N_EDGES;

    char* base = (char*)d_ws;
    size_t off = 0;
    auto alloc = [&](size_t bytes) -> void* {
        void* p = base + off;
        off += (bytes + 255) & ~(size_t)255;
        return p;
    };
    float* dinv = (float*)alloc((size_t)N_NODES * 4);
    bf16*  xb   = (bf16*) alloc((size_t)N_NODES * IN_C * 2);
    bf16*  w1t  = (bf16*) alloc((size_t)IN_C * HID_C * 2);
    bf16*  w2t  = (bf16*) alloc((size_t)HID_C * OUT_C * 2);
    bf16*  h1b  = (bf16*) alloc((size_t)N_NODES * HID_C * 2);
    float* agg1 = (float*)alloc((size_t)N_NODES * HID_C * 4);
    float* h2   = (float*)alloc((size_t)N_NODES * OUT_C * 4);
    float* out  = (float*)d_out;   // 50000 x 128, doubles as agg2

    const int T = 256;
    auto blk = [](long long n, int t) { return (unsigned)((n + t - 1) / t); };

    // normalization
    k_init_deg  <<<blk(N_NODES, T), T, 0, stream>>>(dinv, N_NODES);
    k_accum_deg <<<blk(N_EDGES, T), T, 0, stream>>>(dst, dinv, N_EDGES);
    k_finish_deg<<<blk(N_NODES, T), T, 0, stream>>>(dinv, N_NODES);

    // bf16 conversions
    k_cvt_f32_bf16<<<blk((long long)N_NODES * IN_C / 4, T), T, 0, stream>>>(
        x, xb, (long long)N_NODES * IN_C / 4);
    k_transpose_bf16<<<blk(IN_C * HID_C, T), T, 0, stream>>>(W1, w1t, IN_C, HID_C);
    k_transpose_bf16<<<blk(HID_C * OUT_C, T), T, 0, stream>>>(W2, w2t, HID_C, OUT_C);

    // ---- layer 1 ----
    dim3 g1(blk(N_NODES, 128), HID_C / 64);
    k_gemm_bf16_wmma<<<g1, 256, 0, stream>>>(xb, w1t, nullptr, h1b, N_NODES, HID_C, IN_C);
    k_zero_f32<<<blk((long long)N_NODES * HID_C / 4, T), T, 0, stream>>>(
        agg1, (long long)N_NODES * HID_C / 4);
    k_scatter<<<blk((long long)N_EDGES * (HID_C / 8), T), T, 0, stream>>>(
        src, dst, dinv, h1b, agg1, N_EDGES, HID_C, 5);
    k_final_relu<<<blk((long long)N_NODES * HID_C, T), T, 0, stream>>>(
        agg1, h1b, b1, dinv, HID_C, 8, (long long)N_NODES * HID_C);

    // ---- layer 2 ----
    dim3 g2(blk(N_NODES, 128), OUT_C / 64);
    k_gemm_bf16_wmma<<<g2, 256, 0, stream>>>(h1b, w2t, h2, nullptr, N_NODES, OUT_C, HID_C);
    k_zero_f32<<<blk((long long)N_NODES * OUT_C / 4, T), T, 0, stream>>>(
        out, (long long)N_NODES * OUT_C / 4);
    k_scatter<<<blk((long long)N_EDGES * (OUT_C / 8), T), T, 0, stream>>>(
        src, dst, dinv, h1b, out, N_EDGES, OUT_C, 4);
    k_final_out<<<blk((long long)N_NODES * OUT_C, T), T, 0, stream>>>(
        out, h2, b2, dinv, OUT_C, 7, (long long)N_NODES * OUT_C);
}
